// NaiveLSTMCell_16269336118035
// MI455X (gfx1250) — matrix operand order repaired and checked by reading, not compiled, not run on hardware
//
#include <hip/hip_runtime.h>
#include <hip/hip_bf16.h>
#include <stdint.h>

// ---------------------------------------------------------------------------
// LSTM cell for MI455X (gfx1250, wave32, WMMA + Tensor Data Mover staging).
//   z = [x|h] @ [Wi;Wf;Wo;Wc]^T + b   (B=4096, D=4096, 4H=8192)
//   bf16 WMMA (v_wmma_f32_16x16x32_bf16), f32 accumulate, fused epilogue.
//   B tiles staged into LDS by TDM (tensor_load_to_lds, TENSORcnt), one
//   instruction per 64-wide K slab; triple-buffered, one barrier per slab.
//   Each wave computes 32x32 per gate: every B fragment feeds 2 WMMAs.
// ---------------------------------------------------------------------------

typedef __bf16    v16bf __attribute__((ext_vector_type(16)));
typedef float     v8f   __attribute__((ext_vector_type(8)));
typedef uint32_t  v4u   __attribute__((ext_vector_type(4)));
typedef uint32_t  v8u   __attribute__((ext_vector_type(8)));

#define BATCH 4096
#define INSZ  2048
#define HID   2048
#define DTOT  4096   // INSZ + HID
#define KSLAB 64
#define LROW  72     // padded LDS row (elements): 128B data + 16B pad = 144B

__device__ __forceinline__ uint16_t f2bf(float f) {
  uint32_t b = __float_as_uint(f);
  uint32_t r = (b + 0x7FFFu + ((b >> 16) & 1u)) >> 16;
  return (uint16_t)r;
}

__global__ void pack_xh(const float* __restrict__ x, const float* __restrict__ h,
                        uint16_t* __restrict__ xh) {
  uint32_t gid = (blockIdx.x * blockDim.x + threadIdx.x) * 8u;
  uint32_t row = gid >> 12;
  uint32_t col = gid & 4095u;
  const float* src = (col < INSZ) ? (x + (size_t)row * INSZ + col)
                                  : (h + (size_t)row * HID + (col - INSZ));
  float4 f0 = ((const float4*)src)[0];
  float4 f1 = ((const float4*)src)[1];
  union { uint16_t u[8]; uint4 q; } o;
  o.u[0] = f2bf(f0.x); o.u[1] = f2bf(f0.y); o.u[2] = f2bf(f0.z); o.u[3] = f2bf(f0.w);
  o.u[4] = f2bf(f1.x); o.u[5] = f2bf(f1.y); o.u[6] = f2bf(f1.z); o.u[7] = f2bf(f1.w);
  *(uint4*)(xh + gid) = o.q;
}

__global__ void pack_lin(const float* __restrict__ w, uint16_t* __restrict__ wb) {
  uint32_t gid = (blockIdx.x * blockDim.x + threadIdx.x) * 8u;
  float4 f0 = *(const float4*)(w + gid);
  float4 f1 = *(const float4*)(w + gid + 4);
  union { uint16_t u[8]; uint4 q; } o;
  o.u[0] = f2bf(f0.x); o.u[1] = f2bf(f0.y); o.u[2] = f2bf(f0.z); o.u[3] = f2bf(f0.w);
  o.u[4] = f2bf(f1.x); o.u[5] = f2bf(f1.y); o.u[6] = f2bf(f1.z); o.u[7] = f2bf(f1.w);
  *(uint4*)(wb + gid) = o.q;
}

__device__ __forceinline__ float sigm(float z) {
  return 1.0f / (1.0f + __expf(-z));
}

// Numeric LDS offset of a __shared__ object (AS3 pointers are 32-bit offsets).
__device__ __forceinline__ uint32_t lds_off(const void* p) {
  return (uint32_t)(uintptr_t)(const __attribute__((address_space(3))) void*)p;
}

// ---- Tensor Data Mover: load one B slab (4 gates x 32 rows x 64 k-elems,
// bf16) from W4 into LDS with 16B padding after each 128B row (-> 144B rows).
// D# per CDNA5 ISA (S)8.3-8.6.
__device__ __forceinline__ void tdm_load_slab(uint64_t gaddr, uint32_t lds_byte) {
  v4u g0;
  g0.x = 1u;                                   // count=1, user mode, no gather
  g0.y = lds_byte;                             // lds_addr
  g0.z = (uint32_t)gaddr;                      // global_addr[31:0]
  g0.w = (uint32_t)(gaddr >> 32) | (2u << 30); // global_addr[56:32] | type=2
  v8u g1;
  g1[0] = (1u << 16)      // data_size = 2 bytes
        | (1u << 20)      // pad_enable
        | (4u << 22)      // pad_interval: 32 DWORDs (=128B) between pads
        | (3u << 25);     // pad_amount:   4 DWORDs (=16B)
  g1[1] = (DTOT & 0xFFFFu) << 16;              // tensor_dim0[15:0] (=4096) hi pos
  g1[2] = ((uint32_t)DTOT >> 16) | ((HID & 0xFFFFu) << 16);  // dim0 hi | dim1 lo
  g1[3] = ((uint32_t)HID >> 16) | (KSLAB << 16);             // dim1 hi | tile_dim0=64
  g1[4] = 32u | (4u << 16);                    // tile_dim1=32 | tile_dim2=4
  g1[5] = (uint32_t)DTOT;                      // tensor_dim0_stride lo (=4096)
  g1[6] = 0u;                                  // dim0_stride hi | dim1_stride[15:0]
  g1[7] = (uint32_t)(((uint64_t)HID * DTOT) >> 16);  // dim1_stride[47:16] (=0x80)
  v4u g2;
  g2.x = 4u;                                   // tensor_dim2 = 4 (gates)
  g2.y = 0u;                                   // tensor_dim3 (unused)
  g2.z = (uint32_t)((uint64_t)HID * DTOT);     // tensor_dim2_stride lo
  g2.w = 0u;                                   // stride hi | tile_dim3=0
  v4u g3 = (v4u){0u, 0u, 0u, 0u};              // dims 3/4 unused
  asm volatile("tensor_load_to_lds %0, %1, %2, %3"
               :: "s"(g0), "s"(g1), "s"(g2), "s"(g3) : "memory");
}

// Fused 4-gate GEMM + LSTM epilogue.
// Block = 256 threads = 8 waves. Wave w: rows m0 = by*256 + w*32 (two 16-row
// A fragments). Block: cols n0 = bx*32 (per gate). K loop: slabs of 64.
__global__ __launch_bounds__(256) void lstm_gemm(
    const uint16_t* __restrict__ XH,   // [BATCH][DTOT] bf16
    const uint16_t* __restrict__ W4,   // [4][HID][DTOT] bf16 (i,f,o,c)
    const float* __restrict__ b_i, const float* __restrict__ b_f,
    const float* __restrict__ b_o, const float* __restrict__ b_c,
    const float* __restrict__ c_prev,  // [BATCH][HID]
    float* __restrict__ out_h, float* __restrict__ out_c) {
  // B-tile staging buffers: [3 slabs][4 gates * 32 cols][LROW elems] bf16 (54KB).
  __shared__ __align__(16) uint16_t smem[3][4 * 32 * LROW];

  const int t    = threadIdx.x;
  const int lane = t & 31;
  const int wv   = t >> 5;
  const int l    = lane & 15;
  const int hi   = lane >> 4;
  const int m0   = blockIdx.y * 256 + wv * 32;
  const int n0   = blockIdx.x * 32;

  const uint32_t sbase[3] = { lds_off(&smem[0][0]), lds_off(&smem[1][0]),
                              lds_off(&smem[2][0]) };
  // Global byte address of this block's W tile start; +128B per K slab.
  const uint64_t gbase = (uint64_t)(uintptr_t)W4 + (uint64_t)n0 * DTOT * 2;

  // A-matrix row pointers: rows m0+l and m0+16+l of XH.
  const uint16_t* arow0 = XH + (size_t)(m0 + l) * DTOT;
  const uint16_t* arow1 = arow0 + (size_t)16 * DTOT;

  v8f acc[2][4][2] = {};  // [m-subtile][gate i,f,o,c][16-col half]

  const int nsteps = DTOT / KSLAB;   // 64
  if (wv == 0) {                     // one TDM op per slab for the whole block
    tdm_load_slab(gbase, sbase[0]);
    tdm_load_slab(gbase + (uint64_t)KSLAB * 2, sbase[1]);
  }

  for (int i = 0; i < nsteps; ++i) {
    if (wv == 0) {
      // TDM ops retire in order: <=1 outstanding means slab i is in LDS.
      if (i + 1 < nsteps) __builtin_amdgcn_s_wait_tensorcnt(1);
      else                __builtin_amdgcn_s_wait_tensorcnt(0);
    }
    // Barrier proves (a) slab i visible to all waves, (b) all waves finished
    // slab i-1, so buffer (i+2)%3 == (i-1)%3 is free to refill.
    __syncthreads();
    if (wv == 0 && i + 2 < nsteps)
      tdm_load_slab(gbase + (uint64_t)(i + 2) * KSLAB * 2, sbase[(i + 2) % 3]);

    const uint16_t* bb = &smem[i % 3][0];
    const int k0 = i * KSLAB;
#pragma unroll
    for (int kb = 0; kb < 2; ++kb) {
      // A fragments (ISA layout): v0-3 <- K = kb*32 + hi*8 + {0..7},
      //                           v4-7 <- K = kb*32 + 16 + hi*8 + {0..7}
      union { v16bf v; uint4 q[2]; } A0, A1;
      const uint16_t* ap0 = arow0 + k0 + kb * 32 + hi * 8;
      const uint16_t* ap1 = arow1 + k0 + kb * 32 + hi * 8;
      A0.q[0] = *(const uint4*)ap0;
      A0.q[1] = *(const uint4*)(ap0 + 16);
      A1.q[0] = *(const uint4*)ap1;
      A1.q[1] = *(const uint4*)(ap1 + 16);

      // Each B fragment feeds two WMMAs (both M sub-tiles): ds:wmma = 1:1.
#pragma unroll
      for (int f = 0; f < 8; ++f) {
        const int g = f >> 1, s = f & 1;
        const uint16_t* p = bb + ((g * 32 + s * 16 + l) * LROW + kb * 32 + hi * 8);
        union { v16bf v; uint4 q[2]; } Bf;
        Bf.q[0] = *(const uint4*)p;
        Bf.q[1] = *(const uint4*)(p + 16);
        acc[0][g][s] = __builtin_amdgcn_wmma_f32_16x16x32_bf16(
            false, A0.v, false, Bf.v, (short)0, acc[0][g][s], false, false);
        acc[1][g][s] = __builtin_amdgcn_wmma_f32_16x16x32_bf16(
            false, A1.v, false, Bf.v, (short)0, acc[1][g][s], false, false);
      }
    }
  }

  // Fused LSTM epilogue. C/D layout: VGPR r, lane (l,hi) -> (M = r + 8*hi, N = l).
#pragma unroll
  for (int mh = 0; mh < 2; ++mh) {
#pragma unroll
    for (int s = 0; s < 2; ++s) {
      const int col = n0 + s * 16 + l;
      const float vbi = b_i[col], vbf = b_f[col], vbo = b_o[col], vbc = b_c[col];
#pragma unroll
      for (int r = 0; r < 8; ++r) {
        const int row = m0 + mh * 16 + hi * 8 + r;
        const size_t idx = (size_t)row * HID + col;
        float zi = acc[mh][0][s][r] + vbi;
        float zf = acc[mh][1][s][r] + vbf;
        float zo = acc[mh][2][s][r] + vbo;
        float zc = acc[mh][3][s][r] + vbc;
        float it = sigm(zi);
        float ft = sigm(zf);
        float ot = sigm(zo);
        float ch = tanhf(zc);
        float ct = ft * c_prev[idx] + it * ch;
        float ht = ot * tanhf(ct);
        out_h[idx] = ht;
        out_c[idx] = ct;
      }
    }
  }
}

extern "C" void kernel_launch(void* const* d_in, const int* in_sizes, int n_in,
                              void* d_out, int out_size, void* d_ws, size_t ws_size,
                              hipStream_t stream) {
  const float* x  = (const float*)d_in[0];
  const float* h  = (const float*)d_in[1];
  const float* c  = (const float*)d_in[2];
  const float* Wi = (const float*)d_in[3];
  const float* bi = (const float*)d_in[4];
  const float* Wf = (const float*)d_in[5];
  const float* bf = (const float*)d_in[6];
  const float* Wo = (const float*)d_in[7];
  const float* bo = (const float*)d_in[8];
  const float* Wc = (const float*)d_in[9];
  const float* bc = (const float*)d_in[10];

  uint16_t* xh = (uint16_t*)d_ws;
  uint16_t* w4 = xh + (size_t)BATCH * DTOT;
  const size_t G = (size_t)HID * DTOT;

  float* out_h = (float*)d_out;
  float* out_c = out_h + (size_t)BATCH * HID;

  pack_xh<<<(BATCH * (size_t)DTOT) / (8 * 256), 256, 0, stream>>>(x, h, xh);
  pack_lin<<<G / (8 * 256), 256, 0, stream>>>(Wi, w4 + 0 * G);
  pack_lin<<<G / (8 * 256), 256, 0, stream>>>(Wf, w4 + 1 * G);
  pack_lin<<<G / (8 * 256), 256, 0, stream>>>(Wo, w4 + 2 * G);
  pack_lin<<<G / (8 * 256), 256, 0, stream>>>(Wc, w4 + 3 * G);

  dim3 grid(HID / 32, BATCH / 256);  // 64 x 16 blocks
  lstm_gemm<<<grid, 256, 0, stream>>>(xh, w4, bi, bf, bo, bc, c, out_h, out_c);
}
